// DepthwiseCrossCorrelation_org_14542759264964
// MI455X (gfx1250) — matrix operand order repaired and checked by reading, not compiled, not run on hardware
//
#include <hip/hip_runtime.h>

// ---- types ----
typedef _Float16 v16h __attribute__((ext_vector_type(16)));
typedef _Float16 v8h  __attribute__((ext_vector_type(8)));
typedef _Float16 v4h  __attribute__((ext_vector_type(4)));
typedef float    v8f  __attribute__((ext_vector_type(8)));
typedef float    v4f  __attribute__((ext_vector_type(4)));

// ---- problem constants ----
#define BS_   32
#define NC_   256
#define HI_   64
#define WI_   64
#define HT_   16
#define WT_   16
#define HO_   49
#define WO_   49

// LDS image: zero-padded so A-tile reads (row up to 78, col up to 79) stay in-bounds
#define XROWS  80
#define XPITCH 80   // halves; 160B row stride keeps all v8h loads 16B-aligned

__global__ __launch_bounds__(128)
void dwxcorr_wmma_f16(const float* __restrict__ x,
                      const float* __restrict__ tmpl,
                      float* __restrict__ out) {
    __shared__ __align__(16) _Float16 sX[XROWS * XPITCH];   // 12800 B
    __shared__ __align__(16) _Float16 sT[HT_ * WT_];        //   512 B
    __shared__ __align__(16) _Float16 sB[16 * 16 * 32];     // 16384 B  [ty][n][k]

    const int tid = threadIdx.x;
    const int g   = blockIdx.x;            // group = (bs, nc)

    // ---- 1) zero LDS image (covers padding rows 64..79 / cols 64..79) ----
    {
        uint32_t* p = (uint32_t*)sX;       // 3200 dwords
        #pragma unroll
        for (int i = 0; i < 25; ++i)
            p[tid + i * 128] = 0u;
    }
    __syncthreads();

    // ---- 2) stage X (fp32 -> f16) and template into LDS ----
    {
        const float* xg = x + (size_t)g * (HI_ * WI_);
        #pragma unroll
        for (int i = 0; i < 8; ++i) {
            int idx = tid + i * 128;              // 1024 float4 tiles
            int row = idx >> 4;                   // 16 float4 per row
            int c4  = (idx & 15) << 2;
            v4f f = *(const v4f*)(xg + row * WI_ + c4);
            v4h h;
            h.x = (_Float16)f.x; h.y = (_Float16)f.y;
            h.z = (_Float16)f.z; h.w = (_Float16)f.w;
            *(v4h*)&sX[row * XPITCH + c4] = h;
        }
        if (tid < 64) {
            const float* tg = tmpl + (size_t)g * (HT_ * WT_);
            v4f f = *(const v4f*)(tg + tid * 4);
            v4h h;
            h.x = (_Float16)f.x; h.y = (_Float16)f.y;
            h.z = (_Float16)f.z; h.w = (_Float16)f.w;
            *(v4h*)&sT[tid * 4] = h;
        }
    }
    __syncthreads();

    // ---- 3) build banded B matrices: B_ty[k,n] = T[ty, k-n] if 0<=k-n<16 else 0
    //         stored n-major: sB[ty*512 + n*32 + k] ----
    {
        #pragma unroll
        for (int i = 0; i < 64; ++i) {
            int idx = tid + i * 128;              // 0..8191
            int ty  = idx >> 9;
            int n   = (idx >> 5) & 15;
            int k   = idx & 31;
            int d   = k - n;
            _Float16 v = (d >= 0 && d < 16) ? sT[ty * 16 + d] : (_Float16)0.0f;
            sB[idx] = v;
        }
    }
    __syncthreads();

    // ---- 4) WMMA accumulation: each wave owns one 16-row output band ----
    const int lane = tid & 31;
    const int wv   = tid >> 5;
    const int m    = lane & 15;        // A row / B col / D col within tile
    const int hi   = lane >> 4;        // K-half selector per CDNA5 fragment layout
    const int y0   = wv * 16;          // output row base: 0,16,32,48

    v8f acc0 = {}; v8f acc1 = {}; v8f acc2 = {}; v8f acc3 = {};

    #pragma unroll
    for (int ty = 0; ty < 16; ++ty) {
        // B fragment: lanes 0-15 -> K {0..7,16..23}, lanes 16-31 -> K {8..15,24..31}, col n = m
        const _Float16* bp = &sB[ty * 512 + m * 32 + hi * 8];
        v8h blo = *(const v8h*)bp;
        v8h bh8 = *(const v8h*)(bp + 16);
        v16h bfrag = __builtin_shufflevector(blo, bh8,
            0,1,2,3,4,5,6,7,8,9,10,11,12,13,14,15);

        // A rows: X[y0 + ty + m, *]; K packing: lanes<16 hold {0..7,16..23},
        // lanes>=16 hold {8..15,24..31} relative to tile x-start.
        const int row = y0 + ty + m;
        const _Float16* xrow = &sX[row * XPITCH + hi * 8];

        v8h ch0 = *(const v8h*)(xrow +  0);
        v8h ch1 = *(const v8h*)(xrow + 16);
        v8h ch2 = *(const v8h*)(xrow + 32);
        v8h ch3 = *(const v8h*)(xrow + 48);
        v8h ch4 = *(const v8h*)(xrow + 64);

        v16h a0 = __builtin_shufflevector(ch0, ch1, 0,1,2,3,4,5,6,7,8,9,10,11,12,13,14,15);
        v16h a1 = __builtin_shufflevector(ch1, ch2, 0,1,2,3,4,5,6,7,8,9,10,11,12,13,14,15);
        v16h a2 = __builtin_shufflevector(ch2, ch3, 0,1,2,3,4,5,6,7,8,9,10,11,12,13,14,15);
        v16h a3 = __builtin_shufflevector(ch3, ch4, 0,1,2,3,4,5,6,7,8,9,10,11,12,13,14,15);

        acc0 = __builtin_amdgcn_wmma_f32_16x16x32_f16(false, a0, false, bfrag,
                                                      (short)0, acc0, false, false);
        acc1 = __builtin_amdgcn_wmma_f32_16x16x32_f16(false, a1, false, bfrag,
                                                      (short)0, acc1, false, false);
        acc2 = __builtin_amdgcn_wmma_f32_16x16x32_f16(false, a2, false, bfrag,
                                                      (short)0, acc2, false, false);
        acc3 = __builtin_amdgcn_wmma_f32_16x16x32_f16(false, a3, false, bfrag,
                                                      (short)0, acc3, false, false);
    }

    // ---- 5) store valid 49x49 region; D layout: VGPR i -> row i + 8*hi, col m ----
    float* og = out + (size_t)g * (HO_ * WO_);
    {
        v8f accs[4] = {acc0, acc1, acc2, acc3};
        #pragma unroll
        for (int t = 0; t < 4; ++t) {
            int xcol = t * 16 + m;
            if (xcol < WO_) {
                #pragma unroll
                for (int i = 0; i < 8; ++i) {
                    int yy = y0 + hi * 8 + i;
                    if (yy < HO_)
                        og[yy * WO_ + xcol] = accs[t][i];
                }
            }
        }
    }
}

extern "C" void kernel_launch(void* const* d_in, const int* in_sizes, int n_in,
                              void* d_out, int out_size, void* d_ws, size_t ws_size,
                              hipStream_t stream) {
    const float* x = (const float*)d_in[0];       // [32,256,64,64] fp32
    const float* t = (const float*)d_in[1];       // [32,256,16,16] fp32
    float*     out = (float*)d_out;               // [32,256,49,49] fp32

    dim3 grid(BS_ * NC_);   // 8192 groups, one workgroup each
    dim3 block(128);        // 4 wave32
    hipLaunchKernelGGL(dwxcorr_wmma_f16, grid, block, 0, stream, x, t, out);
}
